// StoX_Conv2d_72335839199774
// MI455X (gfx1250) — 8-bit matrix operand order repaired and checked by reading, compiled, not run on hardware
//
#include <hip/hip_runtime.h>
#include <hip/hip_bf16.h>

typedef __attribute__((ext_vector_type(8))) int v8i;

#define OFF_QA   ((size_t)0)
#define SZ_QA    ((size_t)32768 * 576)
#define OFF_QW   (OFF_QA + SZ_QA)
#define SZ_QW    ((size_t)256 * 576)
#define OFF_PK   (OFF_QW + SZ_QW)
#define SZ_PK    ((size_t)576 * 1024 * 4)
#define OFF_G    (OFF_PK + SZ_PK)
#define SZ_G     ((size_t)9 * 64 * 64 * 4)
#define OFF_ST   (OFF_G + SZ_G)

__global__ __launch_bounds__(32) void k_qw(const float* __restrict__ wf,
                                           signed char* __restrict__ qw) {
    int o = blockIdx.x;
    const float* row = wf + (size_t)o * 576;
    float s = 0.f;
    for (int i = threadIdx.x; i < 576; i += 32) s += row[i];
    #pragma unroll
    for (int off = 16; off > 0; off >>= 1) s += __shfl_xor(s, off, 32);
    float mu = s * (1.0f / 576.0f);
    for (int i = threadIdx.x; i < 576; i += 32) {
        float d = row[i] - mu;
        qw[(size_t)o * 576 + i] = (signed char)((d > 0.f) - (d < 0.f));
    }
}

__global__ __launch_bounds__(256) void k_qa(const float* __restrict__ in,
                                            signed char* __restrict__ qa) {
    int t = blockIdx.x * 256 + threadIdx.x;
    int b = t >> 16;
    int c = (t >> 10) & 63;
    int l = t & 1023;
    int h = l >> 5, w = l & 31;
    const float* inb = in + ((size_t)(b * 64 + c) << 10);
    #pragma unroll
    for (int tap = 0; tap < 9; tap++) {
        int di = tap / 3 - 1, dj = tap % 3 - 1;
        int hh = h + di, ww = w + dj;
        float v = (hh >= 0 && hh < 32 && ww >= 0 && ww < 32) ? inb[(hh << 5) + ww] : 0.f;
        v = fminf(fmaxf(v, 0.f), 1.f);
        unsigned q = (unsigned)(int)rintf(v * 15.f);
        int cc = c * 9 + tap;
        #pragma unroll
        for (int a = 0; a < 4; a++) {
            size_t r = (size_t)(((a * 8 + b) << 10) + l);
            qa[r * 576 + cc] = (signed char)((q >> a) & 1u);
        }
    }
}

__global__ __launch_bounds__(256) void k_pack(const signed char* __restrict__ qa,
                                              unsigned* __restrict__ pk) {
    int t = blockIdx.x * 256 + threadIdx.x;
    int cc = t >> 10;
    int w  = t & 1023;
    const signed char* base = qa + (size_t)w * 32 * 576 + cc;
    unsigned m = 0;
    #pragma unroll 8
    for (int i = 0; i < 32; i++) m |= ((unsigned)(base[(size_t)i * 576] & 1)) << i;
    pk[t] = m;
}

__global__ __launch_bounds__(256) void k_gram(const unsigned* __restrict__ pk,
                                              int* __restrict__ G) {
    int t = blockIdx.x * 256 + threadIdx.x;
    int chunk = t >> 12;
    int k1 = (t >> 6) & 63;
    int k2 = t & 63;
    const unsigned* p1 = pk + (size_t)(chunk * 64 + k1) * 1024;
    const unsigned* p2 = pk + (size_t)(chunk * 64 + k2) * 1024;
    int acc = 0;
    #pragma unroll 4
    for (int i = 0; i < 1024; i++) acc += __popc(p1[i] & p2[i]);
    G[t] = acc;
}

__global__ __launch_bounds__(256) void k_stats(const int* __restrict__ G,
                                               const signed char* __restrict__ qw,
                                               const float* __restrict__ gamma,
                                               const float* __restrict__ beta,
                                               float* __restrict__ st) {
    int t = blockIdx.x * 256 + threadIdx.x;
    int chunk = t >> 8;
    int o = t & 255;
    const signed char* wrow = qw + (size_t)o * 576 + chunk * 64;
    int wv[64];
    #pragma unroll
    for (int k = 0; k < 64; k++) wv[k] = (int)wrow[k];
    const int* Gc = G + chunk * 4096;
    long long msum = 0, esum = 0;
    for (int k = 0; k < 64; k++) msum += (long long)wv[k] * Gc[k * 64 + k];
    for (int k1 = 0; k1 < 64; k1++) {
        int w1 = wv[k1];
        if (w1 == 0) continue;
        const int* gr = Gc + k1 * 64;
        long long rs = 0;
        for (int k2 = 0; k2 < 64; k2++) rs += (long long)wv[k2] * gr[k2];
        esum += (long long)w1 * rs;
    }
    const double N = 32768.0;
    double mean = (double)msum / N;
    double var  = (double)esum / N - mean * mean;
    double rstd = 1.0 / sqrt(var + 1e-5);
    float scale = gamma[o] * (float)rstd;
    float shift = beta[o] - scale * (float)mean;
    float* sp = st + (size_t)t * 2;
    sp[0] = scale;
    sp[1] = shift;
}

__global__ __launch_bounds__(256) void k_main(const signed char* __restrict__ qa,
                                              const signed char* __restrict__ qw,
                                              const float* __restrict__ st,
                                              float* __restrict__ out) {
    int wave = (blockIdx.x << 3) + (threadIdx.x >> 5);
    int lane = threadIdx.x & 31;
    int b   = wave >> 9;
    int rem = wave & 511;
    int lt  = rem >> 3;
    int ot  = rem & 7;
    int l_base = lt << 4;
    int o2b    = ot << 4;
    int n = lane & 15;
    int h = lane >> 4;

    float acc[8];
    #pragma unroll
    for (int j = 0; j < 8; j++) acc[j] = 0.f;

    for (int chunk = 0; chunk < 9; chunk++) {
        v8i Bf[2];
        float scl[2], shf[2];
        #pragma unroll
        for (int s = 0; s < 2; s++) {
            int o = (s << 7) + o2b + n;
            const signed char* bp = qw + (size_t)o * 576 + chunk * 64 + (h << 3);
            int2 u0 = *(const int2*)(bp);
            int2 u1 = *(const int2*)(bp + 16);
            int2 u2 = *(const int2*)(bp + 32);
            int2 u3 = *(const int2*)(bp + 48);
            v8i B;
            B[0] = u0.x; B[1] = u0.y; B[2] = u1.x; B[3] = u1.y;
            B[4] = u2.x; B[5] = u2.y; B[6] = u3.x; B[7] = u3.y;
            Bf[s] = B;
            const float* sp = st + (size_t)((chunk << 8) + o) * 2;
            scl[s] = sp[0];
            shf[s] = sp[1];
        }
        #pragma unroll
        for (int a = 0; a < 4; a++) {
            size_t r = (size_t)((((a << 3) + b) << 10) + l_base + n);
            const signed char* ap = qa + r * 576 + chunk * 64 + (h << 3);
            int2 t0 = *(const int2*)(ap);
            int2 t1 = *(const int2*)(ap + 16);
            int2 t2 = *(const int2*)(ap + 32);
            int2 t3 = *(const int2*)(ap + 48);
            v8i A;
            A[0] = t0.x; A[1] = t0.y; A[2] = t1.x; A[3] = t1.y;
            A[4] = t2.x; A[5] = t2.y; A[6] = t3.x; A[7] = t3.y;
            #pragma unroll
            for (int s = 0; s < 2; s++) {
                v8i C = {0, 0, 0, 0, 0, 0, 0, 0};
                C = __builtin_amdgcn_wmma_i32_16x16x64_iu8(
                         false, A,  true, Bf[s], C,
                         false,  false);
                float w_as = (float)(1 << (a + s + 1)) * (1.0f / 45.0f);
                #pragma unroll
                for (int j = 0; j < 8; j++) {
                    float bn = scl[s] * (float)C[j] + shf[s];
                    float sg = (float)((bn > 0.f) - (bn < 0.f));
                    acc[j] += w_as * sg;
                }
            }
        }
    }
    int o2 = o2b + n;
    float* dst = out + (((size_t)((b << 7) + o2)) << 10) + l_base + (h << 3);
    float4 r0 = {acc[0], acc[1], acc[2], acc[3]};
    float4 r1 = {acc[4], acc[5], acc[6], acc[7]};
    *(float4*)(dst)     = r0;
    *(float4*)(dst + 4) = r1;
}

extern "C" void kernel_launch(void* const* d_in, const int* in_sizes, int n_in,
                              void* d_out, int out_size, void* d_ws, size_t ws_size,
                              hipStream_t stream) {
    const float* inputs = (const float*)d_in[0];
    const float* weight = (const float*)d_in[1];
    const float* gamma  = (const float*)d_in[2];
    const float* beta   = (const float*)d_in[3];
    float* out = (float*)d_out;

    char* ws = (char*)d_ws;
    signed char* qa = (signed char*)(ws + OFF_QA);
    signed char* qw = (signed char*)(ws + OFF_QW);
    unsigned*    pk = (unsigned*)(ws + OFF_PK);
    int*         G  = (int*)(ws + OFF_G);
    float*       st = (float*)(ws + OFF_ST);

    k_qw   <<<256,  32, 0, stream>>>(weight, qw);
    k_qa   <<<2048, 256, 0, stream>>>(inputs, qa);
    k_pack <<<2304, 256, 0, stream>>>(qa, pk);
    k_gram <<<144,  256, 0, stream>>>(pk, G);
    k_stats<<<9,    256, 0, stream>>>(G, qw, gamma, beta, st);
    k_main <<<512,  256, 0, stream>>>(qa, qw, st, out);
}
